// MinGRU_36919538876568
// MI455X (gfx1250) — matrix operand order repaired and checked by reading, not compiled, not run on hardware
//
#include <hip/hip_runtime.h>

// ---------------------------------------------------------------------------
// MinGRU (L=2, B=8, S=2048, D=H=1024) for gfx1250 (MI455X).
//  - Gate GEMMs on bf16 WMMA (v_wmma_f32_16x16x32_bf16), f32 accumulate.
//  - Log-space Heinsen recurrence via per-channel wave32 shuffle scan.
// ---------------------------------------------------------------------------

typedef __attribute__((ext_vector_type(16))) __bf16 v16bf;
typedef __attribute__((ext_vector_type(8)))  __bf16 v8bf;
typedef __attribute__((ext_vector_type(8)))  float  v8f;

#define LL 2
#define BB 8
#define SS 2048
#define DD 1024
#define HH 1024
#define M_TOT (BB * SS)   // 16384 rows per GEMM

__device__ __forceinline__ unsigned short f32_to_bf16(float f) {
  unsigned u = __float_as_uint(f);
  return (unsigned short)((u + 0x7FFFu + ((u >> 16) & 1u)) >> 16);  // RNE
}

__device__ __forceinline__ float softplusf(float x) {
  // numerically stable softplus: max(x,0) + log1p(exp(-|x|))
  return fmaxf(x, 0.f) + log1pf(__expf(-fabsf(x)));
}

__device__ __forceinline__ float laddexp(float a, float b) {
  float m = fmaxf(a, b);
  return m + __logf(__expf(a - m) + __expf(b - m));
}

// -------------------------- conversion kernels -----------------------------

__global__ void cvt_kernel(const float* __restrict__ s,
                           unsigned short* __restrict__ d, int n) {
  for (int i = blockIdx.x * blockDim.x + threadIdx.x; i < n;
       i += gridDim.x * blockDim.x)
    d[i] = f32_to_bf16(s[i]);
}

// WT[n*DD + k] = bf16(W[n*DD + k]); W is (HH x DD) row-major.
// Kept N-major so WMMA B-fragments load contiguous K per lane.
__global__ void cvtT_kernel(const float* __restrict__ w,
                            unsigned short* __restrict__ wt) {
  int i = blockIdx.x * blockDim.x + threadIdx.x;  // over DD*HH
  int k = i >> 10;          // / HH
  int n = i & (HH - 1);
  wt[n * DD + k] = f32_to_bf16(w[n * DD + k]);
}

// --------------------------- fused gate GEMM -------------------------------
// Each wave: 32x32 tile of BOTH gate matrices (k and pre_h) sharing A frags.
// Block = 8 waves (2 x 4) -> 64 x 128 tile. K loop step 32, 8 WMMAs/step.

union Frag { v16bf v; v8bf h[2]; };

__global__ void __launch_bounds__(256)
gemm_gates_kernel(const unsigned short* __restrict__ X,    // (M_TOT, DD) bf16
                  const unsigned short* __restrict__ WzT,  // (DD, HH) bf16
                  const unsigned short* __restrict__ WhT,  // (DD, HH) bf16
                  const float* __restrict__ bz,
                  const float* __restrict__ bh,
                  float* __restrict__ lcoef,               // (M_TOT, HH)
                  float* __restrict__ lval) {              // (M_TOT, HH)
  const int lane = threadIdx.x & 31;
  const int wave = threadIdx.x >> 5;
  const int wm = wave >> 2;                       // 0..1
  const int wn = wave & 3;                        // 0..3
  const int m_base = blockIdx.x * 64 + wm * 32;
  const int n_base = blockIdx.y * 128 + wn * 32;

  const int l15 = lane & 15;
  const int hi8 = (lane >> 4) << 3;               // 0 or 8 (K / M half-split)

  v8f acc[2][2][2] = {};                          // [mt][nt][z|h]

  for (int k0 = 0; k0 < DD; k0 += 32) {
    // A fragments: 16x32 bf16, ISA layout (lane = M row, kbase 0/8 by half)
    Frag a[2];
#pragma unroll
    for (int mt = 0; mt < 2; ++mt) {
      const unsigned short* q =
          X + (size_t)(m_base + mt * 16 + l15) * DD + k0 + hi8;
      a[mt].h[0] = *(const v8bf*)(q);        // K = kb .. kb+7
      a[mt].h[1] = *(const v8bf*)(q + 16);   // K = kb+16 .. kb+23
    }
    // B fragments: lane = column N, K order same as A (kbase 0/8 by half)
    Frag bzf[2], bhf[2];
#pragma unroll
    for (int nt = 0; nt < 2; ++nt) {
      const unsigned short* qz =
          WzT + (size_t)(n_base + nt * 16 + l15) * DD + k0 + hi8;
      bzf[nt].h[0] = *(const v8bf*)(qz);
      bzf[nt].h[1] = *(const v8bf*)(qz + 16);
      const unsigned short* qh =
          WhT + (size_t)(n_base + nt * 16 + l15) * DD + k0 + hi8;
      bhf[nt].h[0] = *(const v8bf*)(qh);
      bhf[nt].h[1] = *(const v8bf*)(qh + 16);
    }
    if (k0 + 32 < DD) {  // global_prefetch_b8 for the next K block
      __builtin_prefetch(X + (size_t)(m_base + l15) * DD + k0 + 32, 0, 1);
      __builtin_prefetch(WzT + (size_t)(k0 + 32 + lane) * HH + n_base, 0, 1);
      __builtin_prefetch(WhT + (size_t)(k0 + 32 + lane) * HH + n_base, 0, 1);
    }
#pragma unroll
    for (int mt = 0; mt < 2; ++mt)
#pragma unroll
      for (int nt = 0; nt < 2; ++nt) {
        acc[mt][nt][0] = __builtin_amdgcn_wmma_f32_16x16x32_bf16(
            false, a[mt].v, false, bzf[nt].v, (short)0, acc[mt][nt][0],
            false, false);
        acc[mt][nt][1] = __builtin_amdgcn_wmma_f32_16x16x32_bf16(
            false, a[mt].v, false, bhf[nt].v, (short)0, acc[mt][nt][1],
            false, false);
      }
  }

  // Epilogue: bias + log-space gate math, write log_coeffs / log_values.
#pragma unroll
  for (int mt = 0; mt < 2; ++mt)
#pragma unroll
    for (int nt = 0; nt < 2; ++nt) {
      const int n = n_base + nt * 16 + l15;
      const float bzv = bz[n];
      const float bhv = bh[n];
#pragma unroll
      for (int i = 0; i < 8; ++i) {
        int m = m_base + mt * 16 + hi8 + i;   // C layout: VGPR i -> M=i(+8)
        float kz = acc[mt][nt][0][i] + bzv;
        float lz = -softplusf(-kz);           // log sigmoid(k)
        float lc = -softplusf(kz);            // log(1 - sigmoid(k))
        float ph = acc[mt][nt][1][i] + bhv;
        float lg = (ph >= 0.f) ? __logf(ph + 0.5f) : -softplusf(-ph);
        size_t idx = (size_t)m * HH + n;
        lcoef[idx] = lc;
        lval[idx] = lz + lg;
      }
    }
}

// ------------------------------- scan --------------------------------------
// One wave per (b, h) channel. Affine composition in log space:
//   (c1,v1) o (c2,v2) = (c1+c2, logaddexp(c2+v1, v2))
// Hillis-Steele shuffle scan over 32-step chunks, carried state across chunks.

__global__ void __launch_bounds__(256)
scan_kernel(const float* __restrict__ lcoef, const float* __restrict__ lval,
            float* __restrict__ out_f32, unsigned short* __restrict__ out_b16) {
  const int lane = threadIdx.x & 31;
  const int wid = (blockIdx.x * blockDim.x + threadIdx.x) >> 5;  // 0..B*H-1
  const int b = wid >> 10;          // / HH
  const int n = wid & (HH - 1);
  float carry = -0.6931471805599453f;  // log h0 = log 0.5

  for (int c = 0; c < SS; c += 32) {
    size_t idx = (size_t)(b * SS + c + lane) * HH + n;
    float lc = lcoef[idx];
    float lv = lval[idx];
#pragma unroll
    for (int d = 1; d < 32; d <<= 1) {
      float pc = __shfl_up(lc, (unsigned)d, 32);
      float pv = __shfl_up(lv, (unsigned)d, 32);
      if (lane >= d) {
        lv = laddexp(lc + pv, lv);
        lc = lc + pc;
      }
    }
    float lh = laddexp(lc + carry, lv);
    float h = __expf(lh);
    if (out_f32) out_f32[idx] = h;
    else         out_b16[idx] = f32_to_bf16(h);
    carry = __shfl(lh, 31, 32);
  }
}

// ----------------------------- launcher ------------------------------------

extern "C" void kernel_launch(void* const* d_in, const int* in_sizes, int n_in,
                              void* d_out, int out_size, void* d_ws,
                              size_t ws_size, hipStream_t stream) {
  const float* x  = (const float*)d_in[0];   // (B,S,D)
  const float* Wz = (const float*)d_in[1];   // (L,H,D)
  const float* bz = (const float*)d_in[2];   // (L,H)
  const float* Wh = (const float*)d_in[3];   // (L,H,D)
  const float* bh = (const float*)d_in[4];   // (L,H)
  (void)in_sizes; (void)n_in; (void)out_size; (void)ws_size;

  char* ws = (char*)d_ws;
  size_t off = 0;
  auto walloc = [&](size_t bytes) {
    void* p = ws + off;
    off = (off + bytes + 255) & ~(size_t)255;
    return p;
  };
  unsigned short* xb16 = (unsigned short*)walloc((size_t)M_TOT * DD * 2); // 32MB
  float* lc            = (float*)walloc((size_t)M_TOT * HH * 4);         // 64MB
  float* lv            = (float*)walloc((size_t)M_TOT * HH * 4);         // 64MB
  unsigned short* wzT  = (unsigned short*)walloc((size_t)DD * HH * 2);   //  2MB
  unsigned short* whT  = (unsigned short*)walloc((size_t)DD * HH * 2);   //  2MB

  {
    int n = M_TOT * DD;
    cvt_kernel<<<(n + 255) / 256, 256, 0, stream>>>(x, xb16, n);
  }

  dim3 gg(M_TOT / 64, HH / 128);
  const int ncvtT = (DD * HH) / 256;
  const int scan_blocks = (BB * HH * 32) / 256;

  for (int l = 0; l < LL; ++l) {
    cvtT_kernel<<<ncvtT, 256, 0, stream>>>(Wz + (size_t)l * HH * DD, wzT);
    cvtT_kernel<<<ncvtT, 256, 0, stream>>>(Wh + (size_t)l * HH * DD, whT);
    gemm_gates_kernel<<<gg, 256, 0, stream>>>(xb16, wzT, whT,
                                              bz + l * HH, bh + l * HH,
                                              lc, lv);
    if (l == LL - 1)
      scan_kernel<<<scan_blocks, 256, 0, stream>>>(lc, lv, (float*)d_out,
                                                   nullptr);
    else
      scan_kernel<<<scan_blocks, 256, 0, stream>>>(lc, lv, nullptr, xb16);
  }
}
